// KroneLinear_46239617909449
// MI455X (gfx1250) — compile-verified, matrix operand order checked by reading, not run on hardware
//
#include <hip/hip_runtime.h>

typedef __attribute__((ext_vector_type(16))) __bf16 v16bf;
typedef __attribute__((ext_vector_type(8)))  __bf16 v8bf;
typedef __attribute__((ext_vector_type(8)))  float  v8f;
typedef __attribute__((ext_vector_type(4)))  float  v4f;

#define DIM   128
#define PITCH 136   // 272B row stride: b128 lane accesses hit disjoint bank groups
#define SMEM_BYTES (3 * DIM * PITCH * 2)  // Wbs, Was, Tt  (bf16) = 104448 B -> 3 blocks/WGP

// ISA v16bf lane layout for 16-bit A/B (16x32): row/col = lane%16,
// K-runs = half*8 + [0..7] and half*8 + 16 + [0..7] within the 32-wide k-step.

// Fragment from bf16 LDS (two contiguous 16B runs).
__device__ __forceinline__ v16bf load_frag_lds(const __bf16* base) {
    v8bf lo = *(const v8bf*)(base);
    v8bf hi = *(const v8bf*)(base + 16);
    v16bf f;
#pragma unroll
    for (int i = 0; i < 8; ++i) { f[i] = lo[i]; f[i + 8] = hi[i]; }
    return f;
}

// Fragment straight from row-major f32 global memory (four 16B runs), cvt to bf16.
__device__ __forceinline__ v16bf load_frag_global_f32(const float* base) {
    v4f a0 = *(const v4f*)(base);
    v4f a1 = *(const v4f*)(base + 4);
    v4f b0 = *(const v4f*)(base + 16);
    v4f b1 = *(const v4f*)(base + 20);
    v16bf f;
#pragma unroll
    for (int i = 0; i < 4; ++i) {
        f[i]      = (__bf16)a0[i];
        f[i + 4]  = (__bf16)a1[i];
        f[i + 8]  = (__bf16)b0[i];
        f[i + 12] = (__bf16)b1[i];
    }
    return f;
}

__global__ __launch_bounds__(256) void krone_wmma_kernel(
    const float* __restrict__ x,
    const float* __restrict__ wa,
    const float* __restrict__ wb,
    const float* __restrict__ bias,
    float* __restrict__ y)
{
    extern __shared__ __bf16 smem[];
    __bf16* Wbs = smem;                  // Wb row-major (column access of Wb^T)
    __bf16* Was = Wbs + DIM * PITCH;     // Wa row-major
    __bf16* Ts  = Was + DIM * PITCH;     // Tt[o][j] = T[j][o] (transposed intermediate)

    const int n   = blockIdx.x;
    const int tid = threadIdx.x;
    const float* xb = x + (size_t)n * (DIM * DIM);
    float*       yb = y + (size_t)n * (DIM * DIM);

    // ---- stage weights -> LDS as bf16 (L2-resident after first blocks)
    for (int v = tid; v < DIM * DIM / 4; v += 256) {
        const int row = v >> 5;            // 32 float4 per 128-wide row
        const int col = (v & 31) << 2;
        v4f fa = *(const v4f*)(wa + v * 4);
        v4f fb = *(const v4f*)(wb + v * 4);
        __bf16* pa = &Was[row * PITCH + col];
        __bf16* pb = &Wbs[row * PITCH + col];
#pragma unroll
        for (int i = 0; i < 4; ++i) {
            pa[i] = (__bf16)fa[i];
            pb[i] = (__bf16)fb[i];
        }
    }

    const int lane = tid & 31;
    const int wave = tid >> 5;     // tile-row strip owned by this wave
    const int half = lane >> 4;
    const int mrow = lane & 15;

    // A fragments for stage 1 come straight from global x (read exactly once),
    // overlapping with the weight staging above.
    v16bf afrag[4];
#pragma unroll
    for (int ks = 0; ks < 4; ++ks)
        afrag[ks] = load_frag_global_f32(xb + (wave * 16 + mrow) * DIM + ks * 32 + half * 8);

    __syncthreads();   // weights staged

    // ================= stage 1: T = Xn @ Wb^T, stored transposed =================
#pragma unroll
    for (int tn = 0; tn < 8; ++tn) {
        v8f acc = {};
#pragma unroll
        for (int ks = 0; ks < 4; ++ks) {
            // B[k][n] = Wb[n][k]: fixed n = tn*16+mrow, contiguous k -> row-major Wb
            v16bf bfrag = load_frag_lds(&Wbs[(tn * 16 + mrow) * PITCH + ks * 32 + half * 8]);
            acc = __builtin_amdgcn_wmma_f32_16x16x32_bf16(
                false, afrag[ks], false, bfrag, (short)0, acc, false, false);
        }
        // D lane holds col o = tn*16+mrow, rows j = wave*16 + half*8 + r (contiguous)
        v8bf packed;
#pragma unroll
        for (int r = 0; r < 8; ++r) packed[r] = (__bf16)acc[r];
        *(v8bf*)(&Ts[(tn * 16 + mrow) * PITCH + wave * 16 + half * 8]) = packed;
    }

    // Hoist stage-2 A fragments (Wa) before the barrier to overlap LDS latency.
    v16bf wfrag[4];
#pragma unroll
    for (int ks = 0; ks < 4; ++ks)
        wfrag[ks] = load_frag_lds(&Was[(wave * 16 + mrow) * PITCH + ks * 32 + half * 8]);

    __syncthreads();   // Tt complete

    // ================= stage 2: Y = Wa @ T  (+bias) =================
#pragma unroll
    for (int tn = 0; tn < 8; ++tn) {
        v8f acc = {};
#pragma unroll
        for (int ks = 0; ks < 4; ++ks) {
            // B[k=j][n=o]: fixed o = tn*16+mrow, contiguous j -> row-major Tt
            v16bf bfrag = load_frag_lds(&Ts[(tn * 16 + mrow) * PITCH + ks * 32 + half * 8]);
            acc = __builtin_amdgcn_wmma_f32_16x16x32_bf16(
                false, wfrag[ks], false, bfrag, (short)0, acc, false, false);
        }
        const int o = tn * 16 + mrow;
#pragma unroll
        for (int r = 0; r < 8; ++r) {
            const int a   = wave * 16 + half * 8 + r;
            const int idx = a * DIM + o;
            // y is a 256MB write-once stream vs 192MB L2: keep it non-temporal.
            __builtin_nontemporal_store(acc[r] + bias[idx], &yb[idx]);
        }
    }
}

extern "C" void kernel_launch(void* const* d_in, const int* in_sizes, int n_in,
                              void* d_out, int out_size, void* d_ws, size_t ws_size,
                              hipStream_t stream) {
    (void)in_sizes; (void)n_in; (void)out_size; (void)d_ws; (void)ws_size;
    const float* x    = (const float*)d_in[0];
    const float* wa   = (const float*)d_in[1];
    const float* wb   = (const float*)d_in[2];
    const float* bias = (const float*)d_in[3];
    float*       y    = (float*)d_out;

    // 102 KB dynamic LDS per workgroup (> 64 KB default cap; WGP has 320 KB -> 3 blocks/WGP)
    hipFuncSetAttribute((const void*)krone_wmma_kernel,
                        hipFuncAttributeMaxDynamicSharedMemorySize, SMEM_BYTES);

    krone_wmma_kernel<<<4096, 256, SMEM_BYTES, stream>>>(x, wa, wb, bias, y);
}
